// Attention_59081570125056
// MI455X (gfx1250) — matrix operand order repaired and checked by reading, not compiled, not run on hardware
//
#include <hip/hip_runtime.h>
#include <hip/hip_bf16.h>

// ---------------------------------------------------------------------------
// Problem constants (from reference)
// ---------------------------------------------------------------------------
#define BB       512        // batch
#define T_ENC    64         // encoder timesteps
#define IN_D     512
#define HID      512
#define NCLS     96
#define NSTEPS   26
#define GATES4   (4*HID)    // 2048
#define KCAT     (IN_D+HID) // 1024 (context | h)
#define WIH_LD   (IN_D+NCLS) // 608

// ---------------------------------------------------------------------------
// Types
// ---------------------------------------------------------------------------
typedef __bf16        v16bf __attribute__((ext_vector_type(16)));
typedef float         v8f   __attribute__((ext_vector_type(8)));
typedef unsigned int  u32x4 __attribute__((ext_vector_type(4)));
typedef int           i32x4 __attribute__((ext_vector_type(4)));
typedef int           i32x8 __attribute__((ext_vector_type(8)));

__device__ __forceinline__ __bf16 f2bf(float f) { return (__bf16)f; }

// B is stored pre-tiled: for tile (kt = k/32, nt = n/64), a contiguous 4KB
// block laid out [n%64][k%32] — exactly the LDS layout the WMMA B-fragment
// gather wants, so the GEMM's B staging is a 1-D DMA.
__device__ __forceinline__ size_t btile_off(int k, int n, int ntn) {
    return ((size_t)(k >> 5) * ntn + (size_t)(n >> 6)) * 2048
         + (size_t)(n & 63) * 32 + (size_t)(k & 31);
}

// ---------------------------------------------------------------------------
// TDM: issue a tensor_load_to_lds for a 2-D tile of 2^dsz-byte elements.
// D# layout per CDNA5 ISA (count=1, type=2).
// 6-operand builtin: (u32x4 g0, i32x8 g1, i32x4 g2, i32x4 g3, i32x8 g4, cpol)
// Groups 2/3/4 unused for <=2-D tiles -> zeros.
// ---------------------------------------------------------------------------
__device__ __forceinline__ void tdm_load_2d(unsigned ldsOff, const void* gptr,
                                            unsigned tensor_d0, unsigned tensor_d1,
                                            unsigned stride_d0,
                                            unsigned tile_d0, unsigned tile_d1,
                                            unsigned dsz_log2) {
    unsigned long long ga = (unsigned long long)gptr;
    u32x4 g0;
    g0[0] = 1u;                                               // count=1, user descriptor
    g0[1] = ldsOff;                                           // lds_addr (bytes)
    g0[2] = (unsigned)(ga & 0xFFFFFFFFu);                     // global_addr[31:0]
    g0[3] = (unsigned)((ga >> 32) & 0x1FFFFFFu) | (2u << 30); // global_addr[56:32] | type=2
    i32x8 g1;
    g1[0] = (int)(dsz_log2 << 16);                            // wg_mask=0, data_size
    g1[1] = (int)((tensor_d0 & 0xFFFFu) << 16);               // tensor_dim0[15:0]
    g1[2] = (int)((tensor_d0 >> 16) | ((tensor_d1 & 0xFFFFu) << 16));
    g1[3] = (int)((tensor_d1 >> 16) | (tile_d0 << 16));       // tile_dim0
    g1[4] = (int)tile_d1;                                     // tile_dim1, tile_dim2=0
    g1[5] = (int)stride_d0;                                   // tensor_dim0_stride[31:0]
    g1[6] = 0;
    g1[7] = 0;
    i32x4 g2 = {0, 0, 0, 0};
    i32x4 g3 = {0, 0, 0, 0};
    i32x8 g4 = {0, 0, 0, 0, 0, 0, 0, 0};
    __builtin_amdgcn_tensor_load_to_lds(g0, g1, g2, g3, g4, 0);
}

// ---------------------------------------------------------------------------
// Prep kernels
// ---------------------------------------------------------------------------
__global__ void zero_f32(float* p, int n) {
    int i = blockIdx.x * blockDim.x + threadIdx.x;
    if (i < n) p[i] = 0.0f;
}
__global__ void zero_bf16(__bf16* p, int n) {
    int i = blockIdx.x * blockDim.x + threadIdx.x;
    if (i < n) p[i] = __builtin_bit_cast(__bf16, (unsigned short)0);
}
__global__ void convert_to_bf16(const float* __restrict__ in, __bf16* __restrict__ out, int n) {
    int i = blockIdx.x * blockDim.x + threadIdx.x;
    if (i < n) out[i] = f2bf(in[i]);
}
// in: [rowsIn][colsIn] f32; logical B[k][n] = in[n][k]; write pre-tiled bf16.
__global__ void transpose_to_bf16_tiled(const float* __restrict__ in, __bf16* __restrict__ out,
                                        int rowsIn, int colsIn, int ntn) {
    int idx = blockIdx.x * blockDim.x + threadIdx.x;
    if (idx >= rowsIn * colsIn) return;
    int n = idx / colsIn;      // input row
    int k = idx % colsIn;      // input col
    out[btile_off(k, n, ntn)] = f2bf(in[(size_t)n * colsIn + k]);
}
// Wcat[k][n] (k<1024, n<2048): k<512 -> W_ih[n][k] ; else W_hh[n][k-512]; pre-tiled.
__global__ void build_wcat_tiled(const float* __restrict__ W_ih, const float* __restrict__ W_hh,
                                 __bf16* __restrict__ wcat) {
    int idx = blockIdx.x * blockDim.x + threadIdx.x;
    if (idx >= KCAT * GATES4) return;
    int k = idx / GATES4;
    int n = idx % GATES4;
    float v = (k < IN_D) ? W_ih[(size_t)n * WIH_LD + k]
                         : W_hh[(size_t)n * HID + (k - IN_D)];
    wcat[btile_off(k, n, GATES4 / 64)] = f2bf(v);
}

// ---------------------------------------------------------------------------
// bf16-WMMA GEMM, TDM-staged with LDS double buffering:
//   C[M,N] = A[M,K](bf16, row-major) * B[K,N](bf16, pre-tiled) [+bias0+bias1+gather]
// Block tile 128x64, BK=32; 8 waves (4x2), each wave 32x32 via 4 WMMAs.
// Wave 0 issues next-stage TDMs into the ping-pong buffer, then waits
// s_wait_tensorcnt 2 (TENSORcnt retires in order), so DMA overlaps WMMA.
// ---------------------------------------------------------------------------
#define BM 128
#define BN 64
#define BK 32

__global__ __launch_bounds__(256)
void gemm_bf16_wmma(const __bf16* __restrict__ A, int lda,
                    const __bf16* __restrict__ Bt, int ntn,
                    float* __restrict__ C, int ldc,
                    int M, int N, int K,
                    const float* __restrict__ bias0,
                    const float* __restrict__ bias1,
                    const float* __restrict__ gW, int gLd, int gOff,
                    const int* __restrict__ gIdx, int gIdxStride) {
    __shared__ __bf16 As[2][BM][BK];   // 2 x 8 KB
    __shared__ __bf16 Bs[2][BN][BK];   // 2 x 4 KB

    const int tid  = threadIdx.x;
    const int m0   = blockIdx.x * BM;
    const int n0   = blockIdx.y * BN;
    const int wave = tid >> 5;
    const int lane = tid & 31;
    const int wm   = wave >> 1;          // 0..3
    const int wn   = wave & 1;           // 0..1
    const int half = lane >> 4;          // 0/1
    const int lr   = lane & 15;

    const unsigned ldsA0 = (unsigned)(unsigned long long)&As[0][0][0];
    const unsigned ldsA1 = (unsigned)(unsigned long long)&As[1][0][0];
    const unsigned ldsB0 = (unsigned)(unsigned long long)&Bs[0][0][0];
    const unsigned ldsB1 = (unsigned)(unsigned long long)&Bs[1][0][0];

    v8f acc[2][2] = {};

    const int nk = K / BK;

    // prologue: stage 0 into buffer 0
    if (tid < 32) {
        tdm_load_2d(ldsA0, A + (size_t)m0 * lda,
                    (unsigned)K, (unsigned)M, (unsigned)lda, BK, BM, 1);
        tdm_load_2d(ldsB0, Bt + btile_off(0, n0, ntn),
                    512u, 1u, 512u, 512u, 1u, 3);
    }

    for (int ki = 0; ki < nk; ++ki) {
        const int cur = ki & 1;
        if (tid < 32) {
            if (ki + 1 < nk) {
                const int k1 = (ki + 1) * BK;
                tdm_load_2d(cur ? ldsA0 : ldsA1, A + (size_t)m0 * lda + k1,
                            (unsigned)K, (unsigned)M, (unsigned)lda, BK, BM, 1);
                tdm_load_2d(cur ? ldsB0 : ldsB1, Bt + btile_off(k1, n0, ntn),
                            512u, 1u, 512u, 512u, 1u, 3);
                // in-order retirement: current stage's 2 TDMs done, next 2 in flight
                __builtin_amdgcn_s_wait_tensorcnt(2);
            } else {
                __builtin_amdgcn_s_wait_tensorcnt(0);
            }
        }
        __syncthreads();   // current-stage LDS visible to all waves

        // ---- fragments from LDS (layouts per CDNA5 ISA 7.12.2) ----
        v16bf afrag[2], bfrag[2];
#pragma unroll
        for (int mt = 0; mt < 2; ++mt) {
            const __bf16* p = &As[cur][wm * 32 + mt * 16 + lr][0];
            float4* d = (float4*)&afrag[mt];
            d[0] = *(const float4*)(p + half * 8);          // K = 0..7  / 8..15
            d[1] = *(const float4*)(p + 16 + half * 8);     // K = 16..23 / 24..31
        }
#pragma unroll
        for (int nt = 0; nt < 2; ++nt) {
            const __bf16* p = &Bs[cur][wn * 32 + nt * 16 + lr][0];
            float4* d = (float4*)&bfrag[nt];
            d[0] = *(const float4*)(p + half * 8);
            d[1] = *(const float4*)(p + 16 + half * 8);
        }
#pragma unroll
        for (int mt = 0; mt < 2; ++mt)
#pragma unroll
            for (int nt = 0; nt < 2; ++nt)
                acc[mt][nt] = __builtin_amdgcn_wmma_f32_16x16x32_bf16(
                    false, afrag[mt], false, bfrag[nt],
                    (short)0, acc[mt][nt], false, false);
        __syncthreads();   // reads done before this buffer is DMA-overwritten
    }

    // ---- epilogue + store (C/D layout: VGPR r -> M = r + 8*half, N = lr) ----
#pragma unroll
    for (int nt = 0; nt < 2; ++nt) {
        const int colg = n0 + wn * 32 + nt * 16 + lr;
        if (colg >= N) continue;
        float badd = 0.0f;
        if (bias0) badd += bias0[colg];
        if (bias1) badd += bias1[colg];
#pragma unroll
        for (int mt = 0; mt < 2; ++mt) {
#pragma unroll
            for (int r = 0; r < 8; ++r) {
                const int rowg = m0 + wm * 32 + mt * 16 + half * 8 + r;
                float v = acc[mt][nt][r] + badd;
                if (gW) v += gW[(size_t)colg * gLd + gOff + gIdx[(size_t)rowg * gIdxStride]];
                C[(size_t)rowg * ldc + colg] = v;
            }
        }
    }
}

// ---------------------------------------------------------------------------
// Attention: e = tanh(Hproj + prevproj) . w_score ; softmax over T;
// context = alpha . batch_H  -> written (bf16) into xcat[:, 0:512]
// One block (256 thr, 8 waves) per batch row.
// ---------------------------------------------------------------------------
__global__ __launch_bounds__(256)
void attention_kernel(const float* __restrict__ Hproj,
                      const float* __restrict__ prevproj,
                      const float* __restrict__ wscore,
                      const float* __restrict__ batch_H,
                      __bf16* __restrict__ xcat_bf) {
    const int b = blockIdx.x;
    __shared__ float s_prev[HID];
    __shared__ float s_w[HID];
    __shared__ float s_e[T_ENC];
    __shared__ float s_alpha[T_ENC];

    const int t = threadIdx.x;
    for (int i = t; i < HID; i += 256) { s_prev[i] = prevproj[(size_t)b * HID + i]; s_w[i] = wscore[i]; }
    __syncthreads();

    const int wave = t >> 5, lane = t & 31;
    for (int tt = wave; tt < T_ENC; tt += 8) {
        const float* hp = Hproj + ((size_t)b * T_ENC + tt) * HID;
        float acc = 0.0f;
        for (int h = lane; h < HID; h += 32)
            acc += tanhf(hp[h] + s_prev[h]) * s_w[h];
#pragma unroll
        for (int off = 16; off > 0; off >>= 1) acc += __shfl_down(acc, off, 32);
        if (lane == 0) s_e[tt] = acc;
    }
    __syncthreads();

    if (wave == 0) {
        float e0 = s_e[lane], e1 = s_e[lane + 32];
        float m = fmaxf(e0, e1);
#pragma unroll
        for (int off = 16; off > 0; off >>= 1) m = fmaxf(m, __shfl_xor(m, off, 32));
        float a0 = __expf(e0 - m), a1 = __expf(e1 - m);
        float ssum = a0 + a1;
#pragma unroll
        for (int off = 16; off > 0; off >>= 1) ssum += __shfl_xor(ssum, off, 32);
        float inv = 1.0f / ssum;
        s_alpha[lane] = a0 * inv;
        s_alpha[lane + 32] = a1 * inv;
    }
    __syncthreads();

    for (int d = t; d < IN_D; d += 256) {
        const float* bh = batch_H + (size_t)b * T_ENC * IN_D + d;
        float acc = 0.0f;
#pragma unroll 8
        for (int tt = 0; tt < T_ENC; ++tt) acc += s_alpha[tt] * bh[(size_t)tt * IN_D];
        xcat_bf[(size_t)b * KCAT + d] = f2bf(acc);
    }
}

// ---------------------------------------------------------------------------
// LSTM pointwise (gate order i,f,g,o). c stays f32; h emitted as bf16 into
// both h_bf (GEMM A operand) and xcat[:, 512:1024].
// ---------------------------------------------------------------------------
__global__ void lstm_pointwise(const float* __restrict__ gates,
                               __bf16* __restrict__ h_bf, float* __restrict__ c,
                               __bf16* __restrict__ xcat_bf) {
    int idx = blockIdx.x * blockDim.x + threadIdx.x;
    if (idx >= BB * HID) return;
    int b = idx >> 9, j = idx & (HID - 1);
    const float* g = gates + (size_t)b * GATES4;
    float ig = 1.0f / (1.0f + __expf(-g[j]));
    float fg = 1.0f / (1.0f + __expf(-g[HID + j]));
    float gg = tanhf(g[2 * HID + j]);
    float og = 1.0f / (1.0f + __expf(-g[3 * HID + j]));
    float cn = fg * c[idx] + ig * gg;
    float hn = og * tanhf(cn);
    c[idx] = cn;
    __bf16 hb = f2bf(hn);
    h_bf[idx] = hb;
    xcat_bf[(size_t)b * KCAT + IN_D + j] = hb;
}

// ---------------------------------------------------------------------------
// Host orchestration
// ---------------------------------------------------------------------------
extern "C" void kernel_launch(void* const* d_in, const int* in_sizes, int n_in,
                              void* d_out, int out_size, void* d_ws, size_t ws_size,
                              hipStream_t stream) {
    const float* batch_H = (const float*)d_in[0];
    const int*   text    = (const int*)  d_in[1];
    const float* W_i2h   = (const float*)d_in[2];
    const float* W_h2h   = (const float*)d_in[3];
    const float* b_h2h   = (const float*)d_in[4];
    const float* w_score = (const float*)d_in[5];
    const float* W_ih    = (const float*)d_in[6];
    const float* b_ih    = (const float*)d_in[7];
    const float* W_hh    = (const float*)d_in[8];
    const float* b_hh    = (const float*)d_in[9];
    const float* W_gen   = (const float*)d_in[10];
    const float* b_gen   = (const float*)d_in[11];
    float* out = (float*)d_out;

    // workspace carve-up (256B aligned)
    char* ws = (char*)d_ws;
    size_t off = 0;
    auto carve = [&](size_t bytes) -> void* {
        void* p = ws + off;
        off = (off + bytes + 255) & ~(size_t)255;
        return p;
    };
    float*  Hproj    = (float*) carve((size_t)BB * T_ENC * HID * 4);   // 64 MB
    __bf16* bh_bf    = (__bf16*)carve((size_t)BB * T_ENC * IN_D * 2);  // 32 MB
    __bf16* Wi2h_t   = (__bf16*)carve((size_t)IN_D * HID * 2);
    __bf16* Wh2h_t   = (__bf16*)carve((size_t)HID * HID * 2);
    __bf16* Wcat     = (__bf16*)carve((size_t)KCAT * GATES4 * 2);
    __bf16* Wgen_t   = (__bf16*)carve((size_t)HID * 128 * 2);          // N padded 96->128
    __bf16* h_bf     = (__bf16*)carve((size_t)BB * HID * 2);
    float*  c        = (float*) carve((size_t)BB * HID * 4);
    float*  prevproj = (float*) carve((size_t)BB * HID * 4);
    __bf16* xcat_bf  = (__bf16*)carve((size_t)BB * KCAT * 2);
    float*  gates    = (float*) carve((size_t)BB * GATES4 * 4);

    // ---- prep: bf16 activation copy + pre-tiled bf16 weights ----
    convert_to_bf16<<<(BB * T_ENC * IN_D + 255) / 256, 256, 0, stream>>>(
        batch_H, bh_bf, BB * T_ENC * IN_D);
    transpose_to_bf16_tiled<<<(HID * IN_D + 255) / 256, 256, 0, stream>>>(
        W_i2h, Wi2h_t, HID, IN_D, HID / 64);
    transpose_to_bf16_tiled<<<(HID * HID + 255) / 256, 256, 0, stream>>>(
        W_h2h, Wh2h_t, HID, HID, HID / 64);
    zero_bf16<<<(HID * 128 + 255) / 256, 256, 0, stream>>>(Wgen_t, HID * 128);
    transpose_to_bf16_tiled<<<(NCLS * HID + 255) / 256, 256, 0, stream>>>(
        W_gen, Wgen_t, NCLS, HID, 128 / 64);
    build_wcat_tiled<<<(KCAT * GATES4 + 255) / 256, 256, 0, stream>>>(W_ih, W_hh, Wcat);
    zero_bf16<<<(BB * HID + 255) / 256, 256, 0, stream>>>(h_bf, BB * HID);
    zero_f32<<<(BB * HID + 255) / 256, 256, 0, stream>>>(c, BB * HID);
    zero_bf16<<<(BB * KCAT + 255) / 256, 256, 0, stream>>>(xcat_bf, BB * KCAT);

    // ---- H_proj = batch_H @ W_i2h^T  (M=32768, N=512, K=512) ----
    gemm_bf16_wmma<<<dim3((BB * T_ENC) / BM, HID / BN), 256, 0, stream>>>(
        bh_bf, IN_D, Wi2h_t, HID / 64, Hproj, HID,
        BB * T_ENC, HID, IN_D,
        nullptr, nullptr, nullptr, 0, 0, nullptr, 0);

    // ---- sequential decode ----
    for (int s = 0; s < NSTEPS; ++s) {
        // prevproj = h @ W_h2h^T + b_h2h   (M=512,N=512,K=512)
        gemm_bf16_wmma<<<dim3(BB / BM, HID / BN), 256, 0, stream>>>(
            h_bf, HID, Wh2h_t, HID / 64, prevproj, HID,
            BB, HID, HID,
            b_h2h, nullptr, nullptr, 0, 0, nullptr, 0);

        // attention -> context (bf16) into xcat[:, :512]
        attention_kernel<<<BB, 256, 0, stream>>>(Hproj, prevproj, w_score, batch_H, xcat_bf);

        // gates = [context|h] @ Wcat + b_ih + b_hh + W_ih[:,512+text[b,s]]
        gemm_bf16_wmma<<<dim3(BB / BM, GATES4 / BN), 256, 0, stream>>>(
            xcat_bf, KCAT, Wcat, GATES4 / 64, gates, GATES4,
            BB, GATES4, KCAT,
            b_ih, b_hh, W_ih, WIH_LD, IN_D, text + s, NSTEPS);

        // LSTM cell pointwise; writes h_bf, c, xcat[:,512:]
        lstm_pointwise<<<(BB * HID + 255) / 256, 256, 0, stream>>>(gates, h_bf, c, xcat_bf);

        // probs[:, s, :] = h @ W_gen^T + b_gen  (M=512, N=96, K=512)
        gemm_bf16_wmma<<<dim3(BB / BM, 2), 256, 0, stream>>>(
            h_bf, HID, Wgen_t, 128 / 64, out + (size_t)s * NCLS, NSTEPS * NCLS,
            BB, NCLS, HID,
            b_gen, nullptr, nullptr, 0, 0, nullptr, 0);
    }
    (void)in_sizes; (void)n_in; (void)out_size; (void)ws_size;
}
